// ScaledDotProductAttention_29618094473464
// MI455X (gfx1250) — compile-verified, matrix operand order checked by reading, not run on hardware
//
#include <hip/hip_runtime.h>

// Flash-attention for B=4, S=4096, D=64 (fp32 in/out), bf16 WMMA compute.
// mask input is the jax bool array (1 byte per element), True = masked key.

#define Bb 4
#define Ss 4096
#define Dd 64
#define BN 64              // keys per main-loop tile
#define NWAVES 4           // waves per block (wave32)
#define BM (NWAVES * 16)   // query rows per block

typedef __attribute__((ext_vector_type(16))) __bf16 v16bf;
typedef __attribute__((ext_vector_type(8)))  __bf16 v8bf;
typedef __attribute__((ext_vector_type(4)))  __bf16 v4bf;
typedef __attribute__((ext_vector_type(8)))  float  v8f;

// Build a 16x16x32 bf16 A/B fragment from two contiguous 8-element (16B) LDS
// groups. Layout per CDNA5 ISA 7.12.2: lane L holds K = base..base+7 in
// elements 0..7 and K = base+16..base+23 in elements 8..15, base = 8*(L/16).
__device__ __forceinline__ v16bf frag16(const __bf16* p0, const __bf16* p1) {
  v8bf lo = *(const v8bf*)p0;
  v8bf hi = *(const v8bf*)p1;
  v16bf r;
#pragma unroll
  for (int i = 0; i < 8; ++i) { r[i] = lo[i]; r[8 + i] = hi[i]; }
  return r;
}

__global__ __launch_bounds__(128)
void fa_fwd_kernel(const float* __restrict__ Q, const float* __restrict__ K,
                   const float* __restrict__ V, const unsigned char* __restrict__ M,
                   float* __restrict__ O) {
  __shared__ __bf16 sK[BN][Dd];          // K tile, [key][dim]   (8 KB)
  __shared__ __bf16 sV[Dd][BN];          // V tile, [dim][key]   (8 KB)
  __shared__ __bf16 sP[NWAVES][16][BN];  // per-wave P scratch   (8 KB)

  const int tid  = threadIdx.x;
  const int wave = tid >> 5;
  const int lane = tid & 31;
  const int half = lane >> 4;   // 0 or 1 (which 16-lane group)
  const int ln   = lane & 15;   // row/col within 16

  const int batch = blockIdx.x / (Ss / BM);
  const int qtile = blockIdx.x % (Ss / BM);
  const int qbase = qtile * BM + wave * 16;

  const unsigned char* mp = M + (size_t)batch * Ss;

  // ---- Q fragments for this wave's 16 rows; fold 1/sqrt(64)=0.125 into cvt
  const float* qrow = Q + ((size_t)batch * Ss + qbase + ln) * Dd;
  v16bf qf[2];
#pragma unroll
  for (int f = 0; f < 2; ++f) {
    const int kb = f * 32 + half * 8;
    const float4* q4 = (const float4*)qrow;
    float4 a0 = q4[kb / 4], a1 = q4[kb / 4 + 1];
    float4 b0 = q4[(kb + 16) / 4], b1 = q4[(kb + 16) / 4 + 1];
    float t0[8] = {a0.x, a0.y, a0.z, a0.w, a1.x, a1.y, a1.z, a1.w};
    float t1[8] = {b0.x, b0.y, b0.z, b0.w, b1.x, b1.y, b1.z, b1.w};
#pragma unroll
    for (int i = 0; i < 8; ++i) {
      qf[f][i]     = (__bf16)(t0[i] * 0.125f);
      qf[f][8 + i] = (__bf16)(t1[i] * 0.125f);
    }
  }

  // ---- online-softmax state: row r of this lane maps to query row r+8*half
  float m_i[8], l_i[8];
  v8f   of[4];  // O accumulators, 4 dim-tiles of 16
#pragma unroll
  for (int r = 0; r < 8; ++r) { m_i[r] = -__builtin_inff(); l_i[r] = 0.f; }
#pragma unroll
  for (int nt = 0; nt < 4; ++nt)
#pragma unroll
    for (int r = 0; r < 8; ++r) of[nt][r] = 0.f;

  const float* kbatch = K + (size_t)batch * Ss * Dd;
  const float* vbatch = V + (size_t)batch * Ss * Dd;

  for (int t = 0; t < Ss / BN; ++t) {
    __syncthreads();  // previous tile fully consumed
    // ---- cooperative K/V tile load: 64x64 fp32 -> bf16 LDS (V transposed)
    const float4* k4 = (const float4*)(kbatch + (size_t)t * BN * Dd);
    const float4* v4 = (const float4*)(vbatch + (size_t)t * BN * Dd);
#pragma unroll
    for (int i = 0; i < 8; ++i) {
      int j = tid + i * 128;        // 0..1023 float4s
      int key = j >> 4, f4 = j & 15;
      float4 kd = k4[j];
      v4bf ks;
      ks[0] = (__bf16)kd.x; ks[1] = (__bf16)kd.y;
      ks[2] = (__bf16)kd.z; ks[3] = (__bf16)kd.w;
      *(v4bf*)&sK[key][f4 * 4] = ks;
      float4 vd = v4[j];
      sV[f4 * 4 + 0][key] = (__bf16)vd.x;
      sV[f4 * 4 + 1][key] = (__bf16)vd.y;
      sV[f4 * 4 + 2][key] = (__bf16)vd.z;
      sV[f4 * 4 + 3][key] = (__bf16)vd.w;
    }
    __syncthreads();

    // ---- S = Q K^T for 4 subtiles of 16 keys (2 WMMAs each, K=64=2x32)
    v8f sf[4];
#pragma unroll
    for (int st = 0; st < 4; ++st) {
      const __bf16* krow = &sK[st * 16 + ln][0];   // B: k=dim, n=key
      v16bf kb0 = frag16(krow + half * 8,      krow + half * 8 + 16);
      v16bf kb1 = frag16(krow + 32 + half * 8, krow + 32 + half * 8 + 16);
      v8f c = {};
      c = __builtin_amdgcn_wmma_f32_16x16x32_bf16(false, qf[0], false, kb0,
                                                  (short)0, c, false, false);
      c = __builtin_amdgcn_wmma_f32_16x16x32_bf16(false, qf[1], false, kb1,
                                                  (short)0, c, false, false);
      sf[st] = c;
    }

    // ---- row max over 64 keys (pre-mask, matching reference), rescale state
#pragma unroll
    for (int r = 0; r < 8; ++r) {
      float v0 = fmaxf(fmaxf(sf[0][r], sf[1][r]), fmaxf(sf[2][r], sf[3][r]));
#pragma unroll
      for (int d = 1; d < 16; d <<= 1) v0 = fmaxf(v0, __shfl_xor(v0, d, 32));
      float mn = fmaxf(m_i[r], v0);
      float sc = __expf(m_i[r] - mn);
      m_i[r] = mn;
      l_i[r] *= sc;
#pragma unroll
      for (int nt = 0; nt < 4; ++nt) of[nt][r] *= sc;
    }

    // ---- P = exp(S - m) * (1 - mask); accumulate row sums; stash bf16 P
    float psum[8];
#pragma unroll
    for (int r = 0; r < 8; ++r) psum[r] = 0.f;
#pragma unroll
    for (int st = 0; st < 4; ++st) {
      int key = t * BN + st * 16 + ln;
      float keep = mp[key] ? 0.f : 1.f;
#pragma unroll
      for (int r = 0; r < 8; ++r) {
        float p = __expf(sf[st][r] - m_i[r]) * keep;
        psum[r] += p;
        sP[wave][r + half * 8][st * 16 + ln] = (__bf16)p;  // C->A via LDS
      }
    }
#pragma unroll
    for (int r = 0; r < 8; ++r) {
      float s = psum[r];
#pragma unroll
      for (int d = 1; d < 16; d <<= 1) s += __shfl_xor(s, d, 32);
      l_i[r] += s;
    }

    // ---- O += P @ V   (LDS ops in-order within the wave; no barrier needed)
#pragma unroll
    for (int kc = 0; kc < 2; ++kc) {
      const __bf16* prow = &sP[wave][ln][0];       // A: m=q-row, k=key
      v16bf pa = frag16(prow + kc * 32 + half * 8,
                        prow + kc * 32 + half * 8 + 16);
#pragma unroll
      for (int nt = 0; nt < 4; ++nt) {
        const __bf16* vrow = &sV[nt * 16 + ln][0]; // B: k=key, n=dim
        v16bf vb = frag16(vrow + kc * 32 + half * 8,
                          vrow + kc * 32 + half * 8 + 16);
        of[nt] = __builtin_amdgcn_wmma_f32_16x16x32_bf16(
            false, pa, false, vb, (short)0, of[nt], false, false);
      }
    }
  }

  // ---- normalize and write out (fp32)
  float* orow = O + ((size_t)batch * Ss + qbase) * Dd;
#pragma unroll
  for (int r = 0; r < 8; ++r) {
    float inv = 1.f / l_i[r];
#pragma unroll
    for (int nt = 0; nt < 4; ++nt)
      orow[(size_t)(r + half * 8) * Dd + nt * 16 + ln] = of[nt][r] * inv;
  }
}

extern "C" void kernel_launch(void* const* d_in, const int* in_sizes, int n_in,
                              void* d_out, int out_size, void* d_ws, size_t ws_size,
                              hipStream_t stream) {
  (void)in_sizes; (void)n_in; (void)out_size; (void)d_ws; (void)ws_size;
  const float* q = (const float*)d_in[0];
  const float* k = (const float*)d_in[1];
  const float* v = (const float*)d_in[2];
  const unsigned char* m = (const unsigned char*)d_in[3];  // jax bool: 1B/elem
  float* out = (float*)d_out;

  dim3 grid(Bb * Ss / BM);   // 256 blocks
  dim3 block(128);           // 4 waves of 32
  fa_fwd_kernel<<<grid, block, 0, stream>>>(q, k, v, m, out);
}
